// AttentionHead_80401787781393
// MI455X (gfx1250) — matrix operand order repaired and checked by reading, not compiled, not run on hardware
//
#include <hip/hip_runtime.h>
#include <hip/hip_bf16.h>

typedef __attribute__((ext_vector_type(16))) __bf16 v16bf;
typedef __attribute__((ext_vector_type(8)))  __bf16 v8bf;
typedef __attribute__((ext_vector_type(8)))  float  v8f;

#define BATCH 8
#define SEQ   1024
#define DMODEL 1024
#define HS    64
#define MAXREL 1024

// ---------------- fragment load helpers (wave32 WMMA bf16 layouts) ----------------
// A-matrix 16x32 bf16: lane l (row = l&15): halves 0-15 = K[kb..kb+15],
// kb = k0 + 16*(l>>4).  Source must be row-major, K contiguous.
__device__ inline v16bf load_a_frag(const __bf16* rowp, int kb) {
    union { v16bf v; v8bf h[2]; } u;
    u.h[0] = *(const v8bf*)(rowp + kb);
    u.h[1] = *(const v8bf*)(rowp + kb + 8);
    return u.v;
}
// B-matrix 32x16 bf16: lane l (col = l&15): halves 0-15 = K[kb..kb+15], kb = k0 + 16*(l>>4).
// Source stored as Bt[n][k], k contiguous.
__device__ inline v16bf load_b_frag(const __bf16* p) {
    union { v16bf v; v8bf h[2]; } u;
    u.h[0] = *(const v8bf*)(p);
    u.h[1] = *(const v8bf*)(p + 8);
    return u.v;
}

// ---------------- pack kernels ----------------
__global__ void pack_f32_bf16(const float* __restrict__ in, __bf16* __restrict__ out, int n) {
    int i = blockIdx.x * 256 + threadIdx.x;
    if (i < n) out[i] = (__bf16)in[i];
}

// Wt[n][k] = W[k][n];  W is [K, N] row-major, Wt is [N, K] row-major
__global__ void pack_w_transpose(const float* __restrict__ W, __bf16* __restrict__ Wt, int K, int N) {
    int i = blockIdx.x * 256 + threadIdx.x;
    if (i < N * K) {
        int n = i / K, k = i - n * K;
        Wt[i] = (__bf16)W[(size_t)k * N + n];
    }
}

// EqT[d][k] = E[1024-d][k] ; EkT[d][k] = E[1024+d][k]   (d in [0,1024), k in [0,64))
__global__ void pack_rel(const float* __restrict__ E, __bf16* __restrict__ EqT, __bf16* __restrict__ EkT) {
    int i = blockIdx.x * 256 + threadIdx.x;
    if (i < MAXREL * HS) {
        int d = i >> 6, k = i & 63;
        EqT[i] = (__bf16)E[(size_t)(MAXREL - d) * HS + k];
        EkT[i] = (__bf16)E[(size_t)(MAXREL + d) * HS + k];
    }
}

// Cd[d] = E[1024+d] . E[1024-d]
__global__ void build_cd(const float* __restrict__ E, float* __restrict__ Cd) {
    int d = blockIdx.x * 256 + threadIdx.x;
    if (d < MAXREL) {
        float s = 0.f;
        for (int k = 0; k < HS; ++k)
            s += E[(size_t)(MAXREL + d) * HS + k] * E[(size_t)(MAXREL - d) * HS + k];
        Cd[d] = s;
    }
}

// ---------------- bf16 WMMA GEMM, 4-way N register blocking ----------------
// Each wave computes a 16x64 output strip: one A fragment feeds 4 B fragments / 4 WMMAs per k-step.
// A row-major [M,K]; Bt stored [N,K] (k contiguous). N must be a multiple of 64.
// vstore==1 (requires N==64): store transposed per batch: out[(b*64 + n)*1024 + t], b = row>>10.
__global__ __launch_bounds__(256) void gemm_bf16_wmma(
    const __bf16* __restrict__ A, const __bf16* __restrict__ Bt,
    const float* __restrict__ bias, __bf16* __restrict__ out,
    int M, int N, int K, int vstore)
{
    const int wave = threadIdx.x >> 5;
    const int lane = threadIdx.x & 31;
    const int lhi = lane >> 4, llo = lane & 15;
    const int ngrp = N >> 6;                       // 64-col groups
    long gid = (long)blockIdx.x * 8 + wave;
    if (gid >= (long)(M >> 4) * ngrp) return;      // uniform per wave
    const int tm = (int)(gid / ngrp);
    const int n0 = (int)(gid - (long)tm * ngrp) * 64;

    v8f acc[4];
    #pragma unroll
    for (int t = 0; t < 4; ++t)
        #pragma unroll
        for (int e = 0; e < 8; ++e) acc[t][e] = 0.f;

    const __bf16* arow  = A  + (size_t)(tm * 16 + llo) * K;
    const __bf16* brow0 = Bt + (size_t)(n0 + llo) * K;
    for (int k0 = 0; k0 < K; k0 += 32) {
        v16bf af = load_a_frag(arow, k0 + 16 * lhi);       // loaded once, used 4x
        #pragma unroll
        for (int t = 0; t < 4; ++t) {
            v16bf bf = load_b_frag(brow0 + (size_t)t * 16 * K + k0 + 16 * lhi);
            acc[t] = __builtin_amdgcn_wmma_f32_16x16x32_bf16(false, af, false, bf, (short)0, acc[t], false, false);
        }
    }

    if (!vstore) {
        #pragma unroll
        for (int t = 0; t < 4; ++t) {
            const float bv = bias ? bias[n0 + t * 16 + llo] : 0.f;
            #pragma unroll
            for (int e = 0; e < 8; ++e) {
                int Mi = e + 8 * lhi;
                out[(size_t)(tm * 16 + Mi) * N + n0 + t * 16 + llo] = (__bf16)(acc[t][e] + bv);
            }
        }
    } else {
        // rows M = 8*lhi + e are contiguous in t-dim -> one 16B store per accumulator
        int row0 = tm * 16 + 8 * lhi;
        int b = row0 >> 10, t0 = row0 & 1023;
        #pragma unroll
        for (int t = 0; t < 4; ++t) {
            v8bf pack;
            #pragma unroll
            for (int e = 0; e < 8; ++e) pack[e] = (__bf16)acc[t][e];
            *(v8bf*)(out + ((size_t)b * HS + n0 + t * 16 + llo) * SEQ + t0) = pack;
        }
    }
}

// ---------------- fused causal attention: scores + softmax + w@v ----------------
// grid = (64 row-blocks, 8 batches), 256 threads = 8 wave32.
// LDS: sScores[16][1024] f32 | sW[16][1024] bf16 | sAcc[16][64] f32 | sInv[16] f32
__global__ __launch_bounds__(256) void attn_fused(
    const __bf16* __restrict__ qh, const __bf16* __restrict__ kh, const __bf16* __restrict__ vt,
    const __bf16* __restrict__ Sq, const __bf16* __restrict__ Sk, const float* __restrict__ Cd,
    float* __restrict__ out)
{
    extern __shared__ char smem[];
    float*  sScores = (float*)smem;                                   // 64 KB
    __bf16* sW      = (__bf16*)(smem + 16 * SEQ * 4);                 // 32 KB
    float*  sAcc    = (float*)(smem + 16 * SEQ * 4 + 16 * SEQ * 2);   // 4 KB
    float*  sInv    = (float*)((char*)sAcc + 16 * HS * 4);            // 64 B

    const int b  = blockIdx.y;
    const int ti = blockIdx.x;
    const int i0 = ti * 16;
    const int tid  = threadIdx.x;
    const int wave = tid >> 5;
    const int lane = tid & 31;
    const int lhi = lane >> 4, llo = lane & 15;
    const float scale = 0.125f; // 1/sqrt(64)

    for (int p = tid; p < 16 * HS; p += 256) sAcc[p] = 0.f;

    const __bf16* qb  = qh + (size_t)b * SEQ * HS;
    const __bf16* kb_ = kh + (size_t)b * SEQ * HS;
    const __bf16* vb  = vt + (size_t)b * HS * SEQ;
    const __bf16* Sqb = Sq + (size_t)b * SEQ * SEQ;
    const __bf16* Skb = Sk + (size_t)b * SEQ * SEQ;

    // ---- Phase 1: raw score tiles: q.k (WMMA) + Sk[j][i-j] (contiguous per lane) ----
    // q A-fragments are invariant over tj: hoist out of the loop.
    v16bf afq0 = load_a_frag(qb + (size_t)(i0 + llo) * HS, 0  + 16 * lhi);
    v16bf afq1 = load_a_frag(qb + (size_t)(i0 + llo) * HS, 32 + 16 * lhi);
    for (int tj = wave; tj <= ti; tj += 8) {
        const int j0 = tj * 16;
        v8f acc;
        #pragma unroll
        for (int e = 0; e < 8; ++e) acc[e] = 0.f;
        v16bf bf0 = load_b_frag(kb_ + (size_t)(j0 + llo) * HS + 0  + 16 * lhi);
        acc = __builtin_amdgcn_wmma_f32_16x16x32_bf16(false, afq0, false, bf0, (short)0, acc, false, false);
        v16bf bf1 = load_b_frag(kb_ + (size_t)(j0 + llo) * HS + 32 + 16 * lhi);
        acc = __builtin_amdgcn_wmma_f32_16x16x32_bf16(false, afq1, false, bf1, (short)0, acc, false, false);

        const int j = j0 + llo;                      // fixed per lane
        const __bf16* skrow = Skb + (size_t)j * SEQ; // d contiguous over e below
        #pragma unroll
        for (int e = 0; e < 8; ++e) {
            const int Mi = e + 8 * lhi;
            const int d = i0 + Mi - j;
            float val = (d < 0) ? -1e30f : (acc[e] + (float)skrow[d]);
            sScores[Mi * SEQ + j0 + llo] = val;
        }
    }
    __syncthreads();

    // ---- Phase 2: per-row add Sq[i][i-j] + Cd[i-j] (contiguous reversed reads), scale, softmax ----
    const int L    = (ti + 1) * 16;
    const int Lpad = (L + 31) & ~31;
    for (int row = wave; row < 16; row += 8) {
        const int i = i0 + row;
        float* srow = sScores + row * SEQ;
        __bf16* wrow = sW + row * SEQ;
        const __bf16* sqrow = Sqb + (size_t)i * SEQ;
        float m = -1e30f;
        for (int j = lane; j < L; j += 32) {
            float v;
            if (j <= i) {
                const int d = i - j;
                v = (srow[j] + (float)sqrow[d] + Cd[d]) * scale;
            } else {
                v = -1e30f;
            }
            srow[j] = v;
            m = fmaxf(m, v);
        }
        #pragma unroll
        for (int o = 16; o > 0; o >>= 1) m = fmaxf(m, __shfl_xor(m, o, 32));
        float s = 0.f;
        for (int j = lane; j < L; j += 32) {
            float e = __expf(srow[j] - m);
            s += e;
            wrow[j] = (__bf16)e;
        }
        for (int j = L + lane; j < Lpad; j += 32) wrow[j] = (__bf16)0.f;
        #pragma unroll
        for (int o = 16; o > 0; o >>= 1) s += __shfl_xor(s, o, 32);
        if (lane == 0) sInv[row] = 1.f / s;
    }
    __syncthreads();

    // ---- Phase 3: out = (w @ v) * invsum via WMMA, A-fragments from LDS ----
    const int Ksteps = Lpad >> 5;        // <= 32
    const int ntile  = wave & 3;         // head-dim 16-col block
    const int kgrp   = wave >> 2;        // split K across 2 wave groups
    v8f oacc;
    #pragma unroll
    for (int e = 0; e < 8; ++e) oacc[e] = 0.f;
    for (int ks = kgrp; ks < Ksteps; ks += 2) {
        const int kb = ks * 32;
        v16bf af = load_a_frag(sW + llo * SEQ, kb + 16 * lhi);                          // ds_read_b128
        v16bf bf = load_b_frag(vb + (size_t)(ntile * 16 + llo) * SEQ + kb + 16 * lhi); // global b128
        oacc = __builtin_amdgcn_wmma_f32_16x16x32_bf16(false, af, false, bf, (short)0, oacc, false, false);
    }
    #pragma unroll
    for (int e = 0; e < 8; ++e) {
        const int Mi = e + 8 * lhi;
        atomicAdd(&sAcc[Mi * HS + ntile * 16 + llo], oacc[e]);   // ds_add_f32
    }
    __syncthreads();

    float* ob = out + ((size_t)b * SEQ + i0) * HS;
    for (int p = tid; p < 16 * HS; p += 256) {
        const int Mi = p >> 6, hd = p & 63;
        ob[(size_t)Mi * HS + hd] = sAcc[p] * sInv[Mi];
    }
}

// ---------------- host launcher ----------------
extern "C" void kernel_launch(void* const* d_in, const int* in_sizes, int n_in,
                              void* d_out, int out_size, void* d_ws, size_t ws_size,
                              hipStream_t stream) {
    const float* x   = (const float*)d_in[0];
    const float* Wk  = (const float*)d_in[1];
    const float* bk  = (const float*)d_in[2];
    const float* Wq  = (const float*)d_in[3];
    const float* bq  = (const float*)d_in[4];
    const float* Wv  = (const float*)d_in[5];
    const float* rel = (const float*)d_in[6];
    float* out = (float*)d_out;

    // workspace partition (bytes)
    char* p = (char*)d_ws;
    auto bump = [&](size_t bytes) { char* r = p; p += (bytes + 255) & ~(size_t)255; return r; };
    __bf16* xh  = (__bf16*)bump((size_t)BATCH * SEQ * DMODEL * 2);  // 16 MB
    __bf16* qh  = (__bf16*)bump((size_t)BATCH * SEQ * HS * 2);      // 1 MB
    __bf16* kh  = (__bf16*)bump((size_t)BATCH * SEQ * HS * 2);      // 1 MB
    __bf16* vt  = (__bf16*)bump((size_t)BATCH * HS * SEQ * 2);      // 1 MB (transposed)
    __bf16* Wqt = (__bf16*)bump((size_t)DMODEL * HS * 2);
    __bf16* Wkt = (__bf16*)bump((size_t)DMODEL * HS * 2);
    __bf16* Wvt = (__bf16*)bump((size_t)DMODEL * HS * 2);
    __bf16* EqT = (__bf16*)bump((size_t)MAXREL * HS * 2);
    __bf16* EkT = (__bf16*)bump((size_t)MAXREL * HS * 2);
    float*  Cd  = (float*) bump((size_t)MAXREL * 4);
    __bf16* Sq  = (__bf16*)bump((size_t)BATCH * SEQ * SEQ * 2);     // 16 MB
    __bf16* Sk  = (__bf16*)bump((size_t)BATCH * SEQ * SEQ * 2);     // 16 MB

    const int nX = BATCH * SEQ * DMODEL;
    pack_f32_bf16<<<(nX + 255) / 256, 256, 0, stream>>>(x, xh, nX);
    pack_w_transpose<<<(DMODEL * HS + 255) / 256, 256, 0, stream>>>(Wq, Wqt, DMODEL, HS);
    pack_w_transpose<<<(DMODEL * HS + 255) / 256, 256, 0, stream>>>(Wk, Wkt, DMODEL, HS);
    pack_w_transpose<<<(DMODEL * HS + 255) / 256, 256, 0, stream>>>(Wv, Wvt, DMODEL, HS);
    pack_rel<<<(MAXREL * HS + 255) / 256, 256, 0, stream>>>(rel, EqT, EkT);
    build_cd<<<(MAXREL + 255) / 256, 256, 0, stream>>>(rel, Cd);

    // projections: q = x@Wq + bq, k = x@Wk + bk, v = x@Wv (stored transposed)
    const int M = BATCH * SEQ;
    const int projBlocks = ((M / 16) * (HS / 64) + 7) / 8;        // 64
    gemm_bf16_wmma<<<projBlocks, 256, 0, stream>>>(xh, Wqt, bq, qh, M, HS, DMODEL, 0);
    gemm_bf16_wmma<<<projBlocks, 256, 0, stream>>>(xh, Wkt, bk, kh, M, HS, DMODEL, 0);
    gemm_bf16_wmma<<<projBlocks, 256, 0, stream>>>(xh, Wvt, nullptr, vt, M, HS, DMODEL, 1);

    // relative-position score tables: Sq[b,i,d] = q_i.E[1024-d], Sk[b,j,d] = k_j.E[1024+d]
    const int relBlocks = ((M / 16) * (MAXREL / 64) + 7) / 8;     // 1024
    gemm_bf16_wmma<<<relBlocks, 256, 0, stream>>>(qh, EqT, nullptr, Sq, M, MAXREL, HS, 0);
    gemm_bf16_wmma<<<relBlocks, 256, 0, stream>>>(kh, EkT, nullptr, Sk, M, MAXREL, HS, 0);

    // fused causal attention
    const size_t shmem = (size_t)16 * SEQ * 4 + (size_t)16 * SEQ * 2 + (size_t)16 * HS * 4 + 64;
    attn_fused<<<dim3(SEQ / 16, BATCH), 256, shmem, stream>>>(qh, kh, vt, Sq, Sk, Cd, out);
}
